// RNNGraphConvModule_75342316306450
// MI455X (gfx1250) — matrix operand order repaired and checked by reading, not compiled, not run on hardware
//
#include <hip/hip_runtime.h>

// ---------------------------------------------------------------------------
// RNN graph-conv (edge-conditioned conv + GRU), MI455X / gfx1250.
// Strategy: fp16 per-edge weight stream (HBM-bound phase), WMMA for the
// filter-net GEMM and the GRU gate matmuls, run-length segmented scatter.
// ---------------------------------------------------------------------------

typedef _Float16 half_t;
typedef __attribute__((ext_vector_type(16))) _Float16 v16h;
typedef __attribute__((ext_vector_type(8)))  _Float16 v8h;
typedef __attribute__((ext_vector_type(2)))  __fp16   fp16v2;  // cvt_pkrtz result type
typedef __attribute__((ext_vector_type(8)))  float    v8f;

#define FE 13

__device__ __forceinline__ v8f wmma_f16(v16h a, v16h b, v8f c) {
  // D = A(16x32 f16) * B(32x16 f16) + C(16x16 f32)
  return __builtin_amdgcn_wmma_f32_16x16x32_f16(false, a, false, b,
                                                (short)0, c, false, false);
}
__device__ __forceinline__ float bcast_f(float v, int srclane) {
  return __int_as_float(__builtin_amdgcn_ds_bpermute(srclane << 2, __float_as_int(v)));
}
__device__ __forceinline__ float sigm_(float x) { return 1.0f / (1.0f + __expf(-x)); }

// ---------------------------------------------------------------- utilities
__global__ void k_zero(float* p, int n) {
  int i = blockIdx.x * blockDim.x + threadIdx.x;
  if (i < n) p[i] = 0.0f;
}

// fp16 transposed copies: W3T[n*128+k] = W3[k*1024+n]; wihT[n*32+k] = wih[k*96+n]
__global__ void k_convert(const float* W3, const float* wih, const float* whh,
                          half_t* W3T, half_t* wihT, half_t* whhT) {
  int i = blockIdx.x * blockDim.x + threadIdx.x;
  if (i < 1024 * 128) {
    int n = i >> 7, k = i & 127;
    W3T[i] = (half_t)W3[k * 1024 + n];
  }
  if (i < 96 * 32) {
    int n = i >> 5, k = i & 31;
    wihT[i] = (half_t)wih[k * 96 + n];
    whhT[i] = (half_t)whh[k * 96 + n];
  }
}

__global__ void k_deg(const int* idxd, float* deg, int E) {
  int e = blockIdx.x * blockDim.x + threadIdx.x;
  if (e < E) atomicAdd(&deg[idxd[e]], 1.0f);
}
__global__ void k_invdeg(const float* deg, float* invdeg, int N) {
  int i = blockIdx.x * blockDim.x + threadIdx.x;
  if (i < N) invdeg[i] = 1.0f / fmaxf(deg[i], 1.0f);
}

// --------------------------------------------- filter net layers 1+2 (VALU)
// h2[e][c] = relu(relu(ef @ W1 + b1) @ W2 + b2), stored fp16 [E][128]
__global__ void k_mlp12(const float* ef, const float* W1, const float* b1,
                        const float* W2, const float* b2, half_t* h2,
                        int E, int ntiles) {
  __shared__ float sW2[32 * 128];
  __shared__ float sEF[4][16 * FE];
  __shared__ float sH1[4][16 * 32];
  int tid = threadIdx.x;
  for (int i = tid; i < 32 * 128; i += 128) sW2[i] = W2[i];
  __syncthreads();
  int w = __builtin_amdgcn_readfirstlane(tid >> 5);
  int lane = tid & 31;
  int tile = blockIdx.x * 4 + w;
  if (tile >= ntiles) return;
  int e0 = tile * 16;
  for (int i = lane; i < 16 * FE; i += 32) {
    int gi = e0 * FE + i;
    sEF[w][i] = (gi < E * FE) ? ef[gi] : 0.0f;
  }
  // layer 1: lane computes h1[t][lane], t = edge-in-tile
  float w1r[FE];
#pragma unroll
  for (int k = 0; k < FE; ++k) w1r[k] = W1[k * 32 + lane];
  float bb1 = b1[lane];
  for (int t = 0; t < 16; ++t) {
    float s = bb1;
#pragma unroll
    for (int k = 0; k < FE; ++k) s += sEF[w][t * FE + k] * w1r[k];
    sH1[w][t * 32 + lane] = fmaxf(s, 0.0f);
  }
  // layer 2: lane computes h2[e][q*32+lane] for q=0..3
  float bb2[4];
#pragma unroll
  for (int q = 0; q < 4; ++q) bb2[q] = b2[q * 32 + lane];
  for (int e = 0; e < 16; ++e) {
    float acc[4] = {bb2[0], bb2[1], bb2[2], bb2[3]};
    for (int k = 0; k < 32; ++k) {
      float hv = sH1[w][e * 32 + k];
#pragma unroll
      for (int q = 0; q < 4; ++q) acc[q] += hv * sW2[k * 128 + q * 32 + lane];
    }
    if (e0 + e < E) {
#pragma unroll
      for (int q = 0; q < 4; ++q)
        h2[(size_t)(e0 + e) * 128 + q * 32 + lane] = (half_t)fmaxf(acc[q], 0.0f);
    }
  }
}

// --------------------------- filter net layer 3 GEMM (WMMA), fp16 output
// weights[e][j][i] = (h2 @ W3 + b3)[e][j*32+i], stored j-pair interleaved:
//   half index = e*1024 + (j>>1)*64 + i*2 + (j&1)
// Each wave: fixed (a,b) -> columns ntA=4a+b (j=2a) and ntB=4a+b+2 (j=2a+1),
// B fragments resident in VGPRs, streams TCHUNK edge tiles.
#define TCHUNK 25

__global__ void k_gemm3(const half_t* h2, const half_t* W3T, const float* b3,
                        unsigned int* wbuf32, int E, int ntiles, int nchunks) {
  int lane = threadIdx.x & 31;
  int wv = __builtin_amdgcn_readfirstlane(threadIdx.x >> 5);  // scalar wave id
  int gw = blockIdx.x * 4 + wv;                               // SGPR
  int g = gw & 31;          // 32 (a,b) groups
  int chunk = gw >> 5;
  if (chunk >= nchunks) return;
  int a = g >> 1, b = g & 1;
  int ntA = a * 4 + b, ntB = ntA + 2;
  int nl = lane & 15, grp = lane >> 4;

  // B fragments: lane holds B[k][n], k = grp*16 + t (contiguous in W3T)
  v16h bA[4], bB[4];
#pragma unroll
  for (int kc = 0; kc < 4; ++kc) {
    bA[kc] = *(const v16h*)(W3T + (ntA * 16 + nl) * 128 + kc * 32 + grp * 16);
    bB[kc] = *(const v16h*)(W3T + (ntB * 16 + nl) * 128 + kc * 32 + grp * 16);
  }
  float biasA = b3[ntA * 16 + nl];
  float biasB = b3[ntB * 16 + nl];
  int cidx = a * 32 + b * 16 + nl;               // packed-pair uint column

  int t0 = chunk * TCHUNK;
  int t1 = min(t0 + TCHUNK, ntiles);
  for (int t = t0; t < t1; ++t) {                // scalar loop
    int ebase = t * 16;                          // SGPR
    // A fragments: lane = row m (edge ebase+nl), k = grp*16 + (0..15)
    int ecl = min(ebase + nl, E - 1);
    const half_t* p = h2 + (size_t)ecl * 128 + grp * 16;
    v16h af[4];
#pragma unroll
    for (int kc = 0; kc < 4; ++kc) {
      v8h lo = *(const v8h*)(p + kc * 32);       // k = kc*32 + grp*16 + (0..7)
      v8h hi = *(const v8h*)(p + kc * 32 + 8);   // k = kc*32 + grp*16 + 8 + (0..7)
      af[kc] = __builtin_shufflevector(lo, hi, 0, 1, 2, 3, 4, 5, 6, 7,
                                       8, 9, 10, 11, 12, 13, 14, 15);
    }
    v8f cA, cB;
#pragma unroll
    for (int r = 0; r < 8; ++r) { cA[r] = biasA; cB[r] = biasB; }
#pragma unroll
    for (int kc = 0; kc < 4; ++kc) {
      cA = wmma_f16(af[kc], bA[kc], cA);
      cB = wmma_f16(af[kc], bB[kc], cB);
    }
    // one base pointer; the 8 stores differ by r*2048 bytes (immediate offsets)
    unsigned int* pst = wbuf32 + (size_t)(ebase + 8 * grp) * 512 + cidx;
    if (ebase + 16 <= E) {                       // scalar fast path, no guards
#pragma unroll
      for (int r = 0; r < 8; ++r) {
        union { half_t h[2]; unsigned int u; } pk;
        pk.h[0] = (half_t)cA[r];                 // j = 2a
        pk.h[1] = (half_t)cB[r];                 // j = 2a+1
        pst[r * 512] = pk.u;
      }
    } else {                                     // tail tile
#pragma unroll
      for (int r = 0; r < 8; ++r) {
        if (ebase + r + 8 * grp < E) {
          union { half_t h[2]; unsigned int u; } pk;
          pk.h[0] = (half_t)cA[r];
          pk.h[1] = (half_t)cB[r];
          pst[r * 512] = pk.u;
        }
      }
    }
  }
}

// ------------------------------- per-edge einsum + segmented scatter (VALU)
// lane = output channel i; acc over sorted-destination runs, atomic at breaks.
// idxd/idxn broadcast via v_readlane -> scalar run test; x broadcast as packed
// half2 (one readlane per j-pair); products via mixed f16->f32 FMA.
__global__ void k_einsum(const float* hin, const unsigned int* wb,
                         const int* idxn, const int* idxd, float* agg, int E) {
  int lane = threadIdx.x & 31;
  int wv = __builtin_amdgcn_readfirstlane(threadIdx.x >> 5);
  int gw = blockIdx.x * 8 + wv;                  // 256-thread blocks = 8 waves
  int e0 = gw * 32;                              // SGPR
  if (e0 >= E) return;
  int cnt = min(32, E - e0);
  int ee = min(e0 + lane, E - 1);
  int dv = idxd[ee];
  int nv = idxn[ee];
  float acc = 0.0f;
  int cur = __builtin_amdgcn_readlane(dv, 0);
  for (int ei = 0; ei < cnt; ++ei) {
    int d = __builtin_amdgcn_readlane(dv, ei);          // uniform (SGPR)
    if (d != cur) {
      atomicAdd(&agg[cur * 32 + lane], acc);
      acc = 0.0f;
      cur = d;
    }
    int node = __builtin_amdgcn_readlane(nv, ei);       // uniform (SGPR)
    float x = hin[(size_t)node * 32 + lane];            // x[lane] = hx[node][lane]
    float xn = bcast_f(x, (lane + 1) & 31);             // neighbor channel
    union { fp16v2 v; int i; } xpk;
    xpk.v = __builtin_amdgcn_cvt_pkrtz(x, xn);          // (x_L, x_L+1) as half2
    const unsigned int* wr = wb + (size_t)(e0 + ei) * 512 + lane;
#pragma unroll
    for (int jj = 0; jj < 16; ++jj) {
      union { unsigned int u; half_t h[2]; } pk;
      pk.u = wr[jj * 32];                               // W[2jj][lane], W[2jj+1][lane]
      union { int i; half_t h[2]; } xb;
      xb.i = __builtin_amdgcn_readlane(xpk.i, 2 * jj);  // (x_2jj, x_2jj+1)
      acc += (float)xb.h[0] * (float)pk.h[0] + (float)xb.h[1] * (float)pk.h[1];
    }
  }
  atomicAdd(&agg[cur * 32 + lane], acc);
}

// ----------------------------------------------- GRU cell (WMMA, 16-node tiles)
__global__ void k_gru(const float* agg, const float* invdeg, const float* hprev,
                      const half_t* wihT, const half_t* whhT,
                      const float* b_ih, const float* b_hh,
                      float* hnext, int N, int ntiles) {
  int lane = threadIdx.x & 31;
  int wv = __builtin_amdgcn_readfirstlane(threadIdx.x >> 5);
  int gw = blockIdx.x * 4 + wv;                  // SGPR tile id
  if (gw >= ntiles) return;
  int nl = lane & 15, grp = lane >> 4;
  int na = min(gw * 16 + nl, N - 1);         // A: m = nl
  float invd = invdeg[na];
  const float* pa = agg + (size_t)na * 32;
  const float* ph = hprev + (size_t)na * 32;
  v16h ax, ah;
#pragma unroll
  for (int t = 0; t < 16; ++t) {
    int k = grp * 16 + t;
    ax[t] = (half_t)(pa[k] * invd);
    ah[t] = (half_t)ph[k];
  }
  v8f gi[6], gh[6];
#pragma unroll
  for (int f = 0; f < 6; ++f) {
    int n = f * 16 + nl;
    v16h bi = *(const v16h*)(wihT + n * 32 + grp * 16);
    v16h bh = *(const v16h*)(whhT + n * 32 + grp * 16);
    float bbi = b_ih[n], bbh = b_hh[n];
    v8f ci, ch;
#pragma unroll
    for (int r = 0; r < 8; ++r) { ci[r] = bbi; ch[r] = bbh; }
    gi[f] = wmma_f16(ax, bi, ci);
    gh[f] = wmma_f16(ah, bh, ch);
  }
  // combine: element (lane,r) of fragment f=gate*2+ct is (m=r+8*grp, col=ct*16+nl)
  const float* phc = hprev + (size_t)(gw * 16 + 8 * grp) * 32;  // + r*32 + col
  float* po = hnext + (size_t)(gw * 16 + 8 * grp) * 32;
  if (gw * 16 + 16 <= N) {                       // scalar full-tile fast path
#pragma unroll
    for (int ct = 0; ct < 2; ++ct) {
      int col = ct * 16 + nl;
#pragma unroll
      for (int r = 0; r < 8; ++r) {
        float hold = phc[r * 32 + col];
        float rg = sigm_(gi[ct][r] + gh[ct][r]);
        float zg = sigm_(gi[2 + ct][r] + gh[2 + ct][r]);
        float ng = tanhf(gi[4 + ct][r] + rg * gh[4 + ct][r]);
        po[r * 32 + col] = (1.0f - zg) * ng + zg * hold;
      }
    }
  } else {                                       // tail tile
#pragma unroll
    for (int ct = 0; ct < 2; ++ct) {
      int col = ct * 16 + nl;
#pragma unroll
      for (int r = 0; r < 8; ++r) {
        int node = gw * 16 + r + 8 * grp;
        float hold = hprev[(size_t)min(node, N - 1) * 32 + col];
        float rg = sigm_(gi[ct][r] + gh[ct][r]);
        float zg = sigm_(gi[2 + ct][r] + gh[2 + ct][r]);
        float ng = tanhf(gi[4 + ct][r] + rg * gh[4 + ct][r]);
        float outv = (1.0f - zg) * ng + zg * hold;
        if (node < N) hnext[(size_t)node * 32 + col] = outv;
      }
    }
  }
}

// ---------------------------------------------------------------------------
extern "C" void kernel_launch(void* const* d_in, const int* in_sizes, int n_in,
                              void* d_out, int out_size, void* d_ws, size_t ws_size,
                              hipStream_t stream) {
  const float* hx  = (const float*)d_in[0];
  const float* ef  = (const float*)d_in[1];
  const int*   idxn = (const int*)d_in[2];
  const int*   idxd = (const int*)d_in[3];
  const float* W1  = (const float*)d_in[4];
  const float* b1  = (const float*)d_in[5];
  const float* W2  = (const float*)d_in[6];
  const float* b2  = (const float*)d_in[7];
  const float* W3  = (const float*)d_in[8];
  const float* b3  = (const float*)d_in[9];
  const float* wih = (const float*)d_in[10];
  const float* whh = (const float*)d_in[11];
  const float* bih = (const float*)d_in[12];
  const float* bhh = (const float*)d_in[13];
  float* out = (float*)d_out;

  const int N = in_sizes[0] / 32;   // 25000
  const int E = in_sizes[2];        // 200000
  const int etiles = (E + 15) / 16;
  const int ntilesN = (N + 15) / 16;

  char* ws = (char*)d_ws;
  size_t off = 0;
  auto alloc = [&](size_t bytes) {
    void* p = ws + off;
    off = (off + bytes + 255) & ~(size_t)255;
    return p;
  };
  half_t* wbuf = (half_t*)alloc((size_t)E * 1024 * sizeof(half_t)); // 410 MB
  half_t* h2   = (half_t*)alloc((size_t)E * 128 * sizeof(half_t));  // 51 MB
  half_t* W3T  = (half_t*)alloc((size_t)1024 * 128 * sizeof(half_t));
  half_t* wihT = (half_t*)alloc((size_t)96 * 32 * sizeof(half_t));
  half_t* whhT = (half_t*)alloc((size_t)96 * 32 * sizeof(half_t));
  float* deg    = (float*)alloc((size_t)N * sizeof(float));
  float* invdeg = (float*)alloc((size_t)N * sizeof(float));
  float* agg    = (float*)alloc((size_t)N * 32 * sizeof(float));
  float* hb0    = (float*)alloc((size_t)N * 32 * sizeof(float));
  float* hb1    = (float*)alloc((size_t)N * 32 * sizeof(float));
  (void)ws_size; (void)n_in; (void)out_size;

  // ---- one-time setup ----
  k_convert<<<(1024 * 128 + 255) / 256, 256, 0, stream>>>(W3, wih, whh, W3T, wihT, whhT);
  k_zero<<<(N + 255) / 256, 256, 0, stream>>>(deg, N);
  k_deg<<<(E + 255) / 256, 256, 0, stream>>>(idxd, deg, E);
  k_invdeg<<<(N + 255) / 256, 256, 0, stream>>>(deg, invdeg, N);
  k_mlp12<<<(etiles + 3) / 4, 128, 0, stream>>>(ef, W1, b1, W2, b2, h2, E, etiles);
  const int nchunks = (etiles + TCHUNK - 1) / TCHUNK;
  const int gemm_waves = 32 * nchunks;
  k_gemm3<<<(gemm_waves + 3) / 4, 128, 0, stream>>>(h2, W3T, b3,
                                                    (unsigned int*)wbuf, E, etiles, nchunks);

  // ---- 10 message-passing + GRU iterations ----
  const float* hin = hx;
  const int ein_waves = (E + 31) / 32;
  for (int it = 0; it < 10; ++it) {
    k_zero<<<(N * 32 + 255) / 256, 256, 0, stream>>>(agg, N * 32);
    k_einsum<<<(ein_waves + 7) / 8, 256, 0, stream>>>(
        hin, (const unsigned int*)wbuf, idxn, idxd, agg, E);
    float* hout = (it == 9) ? out : ((it & 1) ? hb1 : hb0);
    k_gru<<<(ntilesN + 3) / 4, 128, 0, stream>>>(agg, invdeg, hin, wihT, whhT,
                                                 bih, bhh, hout, N, ntilesN);
    hin = hout;
  }
}
